// MultiHeadedAttention_10428180595287
// MI455X (gfx1250) — matrix operand order repaired and checked by reading, not compiled, not run on hardware
//
#include <hip/hip_runtime.h>
#include <hip/hip_bf16.h>

// ---------------------------------------------------------------------------
// MI455X (gfx1250) multi-head attention: bf16 WMMA (f32 accumulate) everywhere,
// with GLOBAL_LOAD_ASYNC_TO_LDS_B128 staging for conversion-free tile copies.
// B=4, S=2048, D=1024, H=16, dk=64.
// ---------------------------------------------------------------------------

typedef __attribute__((ext_vector_type(16))) __bf16 v16bf;
typedef __attribute__((ext_vector_type(8)))  float  v8f;

#define B_   4
#define S_   2048
#define D_   1024
#define H_   16
#define DK_  64
#define EPSF 1e-11f

static __device__ __forceinline__ v8f wmma_bf16(v16bf a, v16bf b, v8f c) {
  // (neg_a, A, neg_b, B, c_mod, C, reuse_a, reuse_b)
  return __builtin_amdgcn_wmma_f32_16x16x32_bf16(false, a, false, b, (short)0, c,
                                                 false, false);
}

// Async DMA: 16 bytes global -> LDS, tracked by ASYNCcnt (no VGPR data path).
static __device__ __forceinline__ void async_copy16(uint32_t lds_byte_addr,
                                                    const void* gaddr) {
  asm volatile("global_load_async_to_lds_b128 %0, %1, off"
               :
               : "v"(lds_byte_addr), "v"(gaddr)
               : "memory");
}
static __device__ __forceinline__ void wait_asynccnt0() {
  asm volatile("s_wait_asynccnt 0x0" ::: "memory");
}

// A-matrix 16x32 bf16 fragment from a row-major tile (stride ld elements).
// Lanes 0-15: row M=lane, K={kb..kb+7, 16+kb..16+kb+7} with kb=(lane>>4)*8.
static __device__ __forceinline__ v16bf load_a_frag(const __bf16* base, int ld,
                                                    int lane) {
  const int row = lane & 15;
  const int kb  = (lane >> 4) * 8;
  const __bf16* p = base + row * ld + kb;
  v16bf f;
#pragma unroll
  for (int j = 0; j < 8; ++j) {
    f[j]     = p[j];
    f[j + 8] = p[16 + j];
  }
  return f;
}

// B-matrix 32x16 bf16 fragment. Source tile stored as [n][k] row-major
// (i.e. B[k][n] = tile[n][k]); lane = output column n, elements sweep K.
static __device__ __forceinline__ v16bf load_b_frag(const __bf16* base, int ld,
                                                    int lane) {
  const int n  = lane & 15;
  const int kb = (lane >> 4) * 8;
  const __bf16* p = base + n * ld + kb;
  v16bf f;
#pragma unroll
  for (int j = 0; j < 8; ++j) {
    f[j]     = p[j];
    f[j + 8] = p[16 + j];
  }
  return f;
}

// ---------------------------------------------------------------------------
// Kernel 1: C = X @ W^T + bias, f32 inputs, bf16 head-major output [B,H,S,dk].
// Block tile 128(M) x 64(N), K-tile 64, 256 threads = 8 waves (wave32),
// each wave computes a 32x32 patch as 2x2 WMMA fragments.
// (f32 -> bf16 conversion required, so staging stays on the VALU path.)
// ---------------------------------------------------------------------------
__global__ __launch_bounds__(256) void proj_qkv(const float* __restrict__ X,
                                                const float* __restrict__ W,
                                                const float* __restrict__ bias,
                                                __bf16* __restrict__ out,
                                                float scale) {
  __shared__ __align__(16) __bf16 As[128][72];
  __shared__ __align__(16) __bf16 Bs[64][72];

  const int tid  = threadIdx.x;
  const int lane = tid & 31;
  const int wid  = tid >> 5;
  const int m0 = blockIdx.y * 128;
  const int n0 = blockIdx.x * 64;
  const int wm = (wid & 3) * 32;
  const int wn = (wid >> 2) * 32;
  const int ln  = lane & 15;
  const int hi8 = (lane >> 4) * 8;

  v8f acc[2][2] = {};

  for (int kk = 0; kk < D_; kk += 64) {
    // Stage A (128x64 f32 -> bf16) into LDS.
    for (int i = tid; i < 128 * 16; i += 256) {
      const int row = i >> 4, c4 = (i & 15) * 4;
      float4 a = *(const float4*)(X + (size_t)(m0 + row) * D_ + kk + c4);
      As[row][c4 + 0] = (__bf16)a.x;
      As[row][c4 + 1] = (__bf16)a.y;
      As[row][c4 + 2] = (__bf16)a.z;
      As[row][c4 + 3] = (__bf16)a.w;
    }
    // Stage W (64x64 f32 -> bf16) into LDS, [n][k] layout.
    for (int i = tid; i < 64 * 16; i += 256) {
      const int row = i >> 4, c4 = (i & 15) * 4;
      float4 b = *(const float4*)(W + (size_t)(n0 + row) * D_ + kk + c4);
      Bs[row][c4 + 0] = (__bf16)b.x;
      Bs[row][c4 + 1] = (__bf16)b.y;
      Bs[row][c4 + 2] = (__bf16)b.z;
      Bs[row][c4 + 3] = (__bf16)b.w;
    }
    __syncthreads();

#pragma unroll
    for (int ks = 0; ks < 64; ks += 32) {
      v16bf a0 = load_a_frag(&As[wm][ks], 72, lane);
      v16bf a1 = load_a_frag(&As[wm + 16][ks], 72, lane);
      v16bf b0 = load_b_frag(&Bs[wn][ks], 72, lane);
      v16bf b1 = load_b_frag(&Bs[wn + 16][ks], 72, lane);
      acc[0][0] = wmma_bf16(a0, b0, acc[0][0]);
      acc[0][1] = wmma_bf16(a0, b1, acc[0][1]);
      acc[1][0] = wmma_bf16(a1, b0, acc[1][0]);
      acc[1][1] = wmma_bf16(a1, b1, acc[1][1]);
    }
    __syncthreads();
  }

  // Epilogue: bias, scale, head-major bf16 store.
#pragma unroll
  for (int i = 0; i < 2; ++i)
#pragma unroll
    for (int j = 0; j < 2; ++j) {
      const int gcol = n0 + wn + j * 16 + ln;
      const float bv = bias[gcol];
      const int hh = gcol >> 6, dd = gcol & 63;
#pragma unroll
      for (int r = 0; r < 8; ++r) {
        const int grow = m0 + wm + i * 16 + r + hi8;
        const int bb = grow >> 11, ss = grow & 2047;
        const float v = (acc[i][j][r] + bv) * scale;
        out[(size_t)((bb * H_ + hh) * S_ + ss) * DK_ + dd] = (__bf16)v;
      }
    }
}

// ---------------------------------------------------------------------------
// Kernel 2: flash attention over one (b,h); 128 query rows per block,
// 8 waves x 16 rows each; 64-key tiles; online softmax with wave32 shuffles.
// K tile staged with async-to-LDS DMA (ASYNCcnt); V staged manually because
// it is transposed in flight for the B-fragment layout.
// ---------------------------------------------------------------------------
__global__ __launch_bounds__(256) void attn_kernel(const __bf16* __restrict__ Q,
                                                   const __bf16* __restrict__ K,
                                                   const __bf16* __restrict__ V,
                                                   const int* __restrict__ mask,
                                                   __bf16* __restrict__ O) {
  __shared__ __align__(16) __bf16 Ks[64][72];      // [key][d]
  __shared__ __align__(16) __bf16 Vt[64][72];      // [d][key]
  __shared__ __align__(16) __bf16 Ps[8][16][72];   // per-wave P scratch

  const int tid  = threadIdx.x;
  const int lane = tid & 31;
  const int wid  = tid >> 5;
  const int bh = blockIdx.y;
  const int b  = bh >> 4;
  const int h  = bh & 15;
  const int q0 = blockIdx.x * 128 + wid * 16;   // seq row base for this wave
  const int ln  = lane & 15;
  const int hi8 = (lane >> 4) * 8;

  const __bf16* qbase = Q + (size_t)bh * S_ * DK_;
  const __bf16* kbase = K + (size_t)bh * S_ * DK_;
  const __bf16* vbase = V + (size_t)bh * S_ * DK_;
  const int*    mbase = mask + (size_t)b * S_ * S_;
  const uint32_t ldsK0 = (uint32_t)(uintptr_t)&Ks[0][0];

  // Q fragments held in registers for the whole pass (A-layout, 2 K-chunks).
  v16bf qa[2];
  {
    const __bf16* qr = qbase + (size_t)(q0 + ln) * DK_;
#pragma unroll
    for (int dc = 0; dc < 2; ++dc)
#pragma unroll
      for (int j = 0; j < 8; ++j) {
        qa[dc][j]     = qr[dc * 32 + hi8 + j];
        qa[dc][j + 8] = qr[dc * 32 + 16 + hi8 + j];
      }
  }

  v8f accO[4] = {};
  float mi[8], li[8];
#pragma unroll
  for (int r = 0; r < 8; ++r) { mi[r] = -1e30f; li[r] = 0.0f; }

  for (int kt = 0; kt < S_; kt += 64) {
    __syncthreads();
    // K tile: async DMA global -> LDS (no VGPR data path, ASYNCcnt-tracked).
    for (int i = tid; i < 64 * 8; i += 256) {
      const int row = i >> 3, c8 = (i & 7) * 8;
      const uint32_t laddr = ldsK0 + (uint32_t)(row * 144 + c8 * 2);
      async_copy16(laddr, kbase + (size_t)(kt + row) * DK_ + c8);
    }
    // V tile: manual transpose into [d][key] for B-fragment loads.
    for (int i = tid; i < 64 * 8; i += 256) {
      const int row = i >> 3, c8 = (i & 7) * 8;
      __bf16 tmp[8];
      *(uint4*)tmp = *(const uint4*)(vbase + (size_t)(kt + row) * DK_ + c8);
#pragma unroll
      for (int j = 0; j < 8; ++j) Vt[c8 + j][row] = tmp[j];
    }
    if (kt + 64 < S_) {
      __builtin_prefetch(kbase + (size_t)(kt + 64) * DK_, 0, 1);
      __builtin_prefetch(vbase + (size_t)(kt + 64) * DK_, 0, 1);
    }
    wait_asynccnt0();
    __syncthreads();

    // scores: 16 rows x 64 keys (4 column subtiles).
    v8f sc[4];
#pragma unroll
    for (int ns = 0; ns < 4; ++ns) {
      v8f c = {};
#pragma unroll
      for (int kc = 0; kc < 2; ++kc) {
        v16bf bk = load_b_frag(&Ks[ns * 16][kc * 32], 72, lane);
        c = wmma_bf16(qa[kc], bk, c);
      }
      sc[ns] = c;
    }

    // Mask (reference: masked score := EPS, still in softmax) + row max.
    float rowmax[8];
#pragma unroll
    for (int r = 0; r < 8; ++r) rowmax[r] = -1e30f;
#pragma unroll
    for (int ns = 0; ns < 4; ++ns) {
      const int gcol = kt + ns * 16 + ln;
#pragma unroll
      for (int r = 0; r < 8; ++r) {
        const int grow = q0 + r + hi8;
        const int mv = mbase[(size_t)grow * S_ + gcol];
        float s = sc[ns][r];
        s = mv ? s : EPSF;
        sc[ns][r] = s;
        rowmax[r] = fmaxf(rowmax[r], s);
      }
    }
#pragma unroll
    for (int r = 0; r < 8; ++r) {
#pragma unroll
      for (int off = 1; off < 16; off <<= 1)
        rowmax[r] = fmaxf(rowmax[r], __shfl_xor(rowmax[r], off, 32));
      const float mn = fmaxf(mi[r], rowmax[r]);
      const float corr = expf(mi[r] - mn);
      mi[r] = mn;
      li[r] *= corr;
#pragma unroll
      for (int ds = 0; ds < 4; ++ds) accO[ds][r] *= corr;
    }

    // P = exp(s - m), row sums, spill P to LDS in bf16 (C-layout -> A-layout).
    float rowsum[8];
#pragma unroll
    for (int r = 0; r < 8; ++r) rowsum[r] = 0.0f;
#pragma unroll
    for (int ns = 0; ns < 4; ++ns)
#pragma unroll
      for (int r = 0; r < 8; ++r) {
        const float p = expf(sc[ns][r] - mi[r]);
        rowsum[r] += p;
        Ps[wid][r + hi8][ns * 16 + ln] = (__bf16)p;
      }
#pragma unroll
    for (int r = 0; r < 8; ++r) {
#pragma unroll
      for (int off = 1; off < 16; off <<= 1)
        rowsum[r] += __shfl_xor(rowsum[r], off, 32);
      li[r] += rowsum[r];
    }

    // O += P @ V  (K dim = 64 keys split into two 32-chunks).
#pragma unroll
    for (int kc = 0; kc < 2; ++kc) {
      v16bf pa = load_a_frag(&Ps[wid][0][kc * 32], 72, lane);
#pragma unroll
      for (int ds = 0; ds < 4; ++ds) {
        v16bf bv = load_b_frag(&Vt[ds * 16][kc * 32], 72, lane);
        accO[ds] = wmma_bf16(pa, bv, accO[ds]);
      }
    }
  }

  // Normalize and store O in row-major [B*S, D] (head-interleaved) bf16.
#pragma unroll
  for (int ds = 0; ds < 4; ++ds)
#pragma unroll
    for (int r = 0; r < 8; ++r) {
      const int grow = q0 + r + hi8;
      const int dd = h * DK_ + ds * 16 + ln;
      const float v = accO[ds][r] / li[r];
      O[((size_t)b * S_ + grow) * D_ + dd] = (__bf16)v;
    }
}

// ---------------------------------------------------------------------------
// Kernel 3: out = O(bf16) @ wo^T + bo, f32 output to d_out.
// A tile is a conversion-free bf16 copy -> async-to-LDS DMA path.
// ---------------------------------------------------------------------------
__global__ __launch_bounds__(256) void out_proj(const __bf16* __restrict__ A,
                                                const float* __restrict__ W,
                                                const float* __restrict__ bias,
                                                float* __restrict__ out) {
  __shared__ __align__(16) __bf16 As[128][72];
  __shared__ __align__(16) __bf16 Bs[64][72];

  const int tid  = threadIdx.x;
  const int lane = tid & 31;
  const int wid  = tid >> 5;
  const int m0 = blockIdx.y * 128;
  const int n0 = blockIdx.x * 64;
  const int wm = (wid & 3) * 32;
  const int wn = (wid >> 2) * 32;
  const int ln  = lane & 15;
  const int hi8 = (lane >> 4) * 8;
  const uint32_t ldsA0 = (uint32_t)(uintptr_t)&As[0][0];

  v8f acc[2][2] = {};

  for (int kk = 0; kk < D_; kk += 64) {
    // A tile: async DMA global(bf16) -> LDS.
    for (int i = tid; i < 128 * 8; i += 256) {
      const int row = i >> 3, c8 = (i & 7) * 8;
      const uint32_t laddr = ldsA0 + (uint32_t)(row * 144 + c8 * 2);
      async_copy16(laddr, A + (size_t)(m0 + row) * D_ + kk + c8);
    }
    // W tile: f32 -> bf16 conversion path.
    for (int i = tid; i < 64 * 16; i += 256) {
      const int row = i >> 4, c4 = (i & 15) * 4;
      float4 b = *(const float4*)(W + (size_t)(n0 + row) * D_ + kk + c4);
      Bs[row][c4 + 0] = (__bf16)b.x;
      Bs[row][c4 + 1] = (__bf16)b.y;
      Bs[row][c4 + 2] = (__bf16)b.z;
      Bs[row][c4 + 3] = (__bf16)b.w;
    }
    wait_asynccnt0();
    __syncthreads();

#pragma unroll
    for (int ks = 0; ks < 64; ks += 32) {
      v16bf a0 = load_a_frag(&As[wm][ks], 72, lane);
      v16bf a1 = load_a_frag(&As[wm + 16][ks], 72, lane);
      v16bf b0 = load_b_frag(&Bs[wn][ks], 72, lane);
      v16bf b1 = load_b_frag(&Bs[wn + 16][ks], 72, lane);
      acc[0][0] = wmma_bf16(a0, b0, acc[0][0]);
      acc[0][1] = wmma_bf16(a0, b1, acc[0][1]);
      acc[1][0] = wmma_bf16(a1, b0, acc[1][0]);
      acc[1][1] = wmma_bf16(a1, b1, acc[1][1]);
    }
    __syncthreads();
  }

#pragma unroll
  for (int i = 0; i < 2; ++i)
#pragma unroll
    for (int j = 0; j < 2; ++j) {
      const int gcol = n0 + wn + j * 16 + ln;
      const float bv = bias[gcol];
#pragma unroll
      for (int r = 0; r < 8; ++r) {
        const int grow = m0 + wm + i * 16 + r + hi8;
        out[(size_t)grow * D_ + gcol] = acc[i][j][r] + bv;
      }
    }
}

// ---------------------------------------------------------------------------
extern "C" void kernel_launch(void* const* d_in, const int* in_sizes, int n_in,
                              void* d_out, int out_size, void* d_ws,
                              size_t ws_size, hipStream_t stream) {
  const float* query = (const float*)d_in[0];
  const float* key   = (const float*)d_in[1];
  const float* value = (const float*)d_in[2];
  const int*   mask  = (const int*)d_in[3];
  const float* wq = (const float*)d_in[4];
  const float* bq = (const float*)d_in[5];
  const float* wk = (const float*)d_in[6];
  const float* bk = (const float*)d_in[7];
  const float* wv = (const float*)d_in[8];
  const float* bv = (const float*)d_in[9];
  const float* wo = (const float*)d_in[10];
  const float* bo = (const float*)d_in[11];
  float* out = (float*)d_out;

  const size_t elems = (size_t)B_ * S_ * D_;   // 8388608
  __bf16* qbf = (__bf16*)d_ws;
  __bf16* kbf = qbf + elems;
  __bf16* vbf = kbf + elems;
  __bf16* obf = vbf + elems;                   // total 64 MB of workspace

  const dim3 gproj(D_ / 64, (B_ * S_) / 128);  // (16, 64)
  const dim3 gattn(S_ / 128, B_ * H_);         // (16, 64)
  const float qscale = 0.125f;                 // 1/sqrt(dk), dk=64

  proj_qkv<<<gproj, 256, 0, stream>>>(query, wq, bq, qbf, qscale);
  proj_qkv<<<gproj, 256, 0, stream>>>(key, wk, bk, kbf, 1.0f);
  proj_qkv<<<gproj, 256, 0, stream>>>(value, wv, bv, vbf, 1.0f);
  attn_kernel<<<gattn, 256, 0, stream>>>(qbf, kbf, vbf, mask, obf);
  out_proj<<<gproj, 256, 0, stream>>>(obf, wo, bo, out);
}
